// GNNEncoder_24919400251990
// MI455X (gfx1250) — compile-verified, matrix operand order checked by reading, not run on hardware
//
#include <hip/hip_runtime.h>
#include <hip/hip_bf16.h>

typedef unsigned short u16;
typedef __attribute__((ext_vector_type(16))) __bf16 v16bf;
typedef __attribute__((ext_vector_type(8)))  float  v8f;

#define NFEAT 128

__device__ __forceinline__ u16 f2bf(float f) {
    unsigned u = __float_as_uint(f);
    unsigned r = (u + 0x7FFFu + ((u >> 16) & 1u)) >> 16;  // RNE f32->bf16
    return (u16)r;
}

union FragU { uint4 q[2]; v16bf v; };

// ---------------- elementwise helpers ----------------
__global__ void zero_f32_kernel(float* p, long n) {
    long i = (long)blockIdx.x * blockDim.x + threadIdx.x;
    if (i < n) p[i] = 0.0f;
}

__global__ void degree_kernel(const int* __restrict__ row, float* __restrict__ deg, int nE) {
    int i = blockIdx.x * blockDim.x + threadIdx.x;
    if (i < nE) atomicAdd(&deg[row[i]], 1.0f);
}

__global__ void dinv_kernel(const float* __restrict__ deg, float* __restrict__ dinv, int n) {
    int i = blockIdx.x * blockDim.x + threadIdx.x;
    if (i < n) {
        float d = deg[i];
        dinv[i] = d > 0.0f ? __frsqrt_rn(d) : 0.0f;
    }
}

__global__ void cvt_bf16_kernel(const float* __restrict__ in, u16* __restrict__ out, long n) {
    long i = (long)blockIdx.x * blockDim.x + threadIdx.x;
    if (i < n) out[i] = f2bf(in[i]);
}

// W[k][n] f32 -> Wt[n][k] bf16 (column-major staging so B-fragments are K-contiguous)
__global__ void wtrans_kernel(const float* __restrict__ W, u16* __restrict__ Wt) {
    int i = blockIdx.x * blockDim.x + threadIdx.x;  // 16384 threads
    int k = i >> 7, n = i & 127;
    Wt[n * NFEAT + k] = f2bf(W[i]);
}

// ---------------- SpMM: txf[row] += -(dinv[row]*dinv[col]) * xsrc[col] ----------------
__global__ __launch_bounds__(256) void spmm_kernel(
    const int* __restrict__ row, const int* __restrict__ col,
    const float* __restrict__ dinv, const float* __restrict__ xsrc,
    float* __restrict__ txf, int nE)
{
    int e = (int)(((long)blockIdx.x * blockDim.x + threadIdx.x) >> 5);  // one wave32 per edge
    int lane = threadIdx.x & 31;
    if (e >= nE) return;
    int r = row[e], c = col[e];
    float w = -(dinv[r] * dinv[c]);
    const float4* xs = (const float4*)(xsrc + (size_t)c * NFEAT);
    float* td = txf + (size_t)r * NFEAT + lane * 4;
    float4 xv = xs[lane];
    atomicAdd(td + 0, xv.x * w);
    atomicAdd(td + 1, xv.y * w);
    atomicAdd(td + 2, xv.z * w);
    atomicAdd(td + 3, xv.w * w);
}

// ---------------- Fused dual-GEMM + bias + ReLU (+ optional average epilogue) ----------
// AVG=false: out = relu(A0@W0 + A1@W1 + b); also writes bf16 copy to outb.
// AVG=true : out = 0.5*(addsrc + relu(A0@W0 + A1@W1 + b)).
// A0/A1: [nrows x 128] bf16 row-major. W0t/W1t: [128 x 128] bf16, N-major (Wt[n][k]).
template <bool AVG>
__global__ __launch_bounds__(256) void gemm_dual_kernel(
    const u16* __restrict__ A0, const u16* __restrict__ A1,
    const u16* __restrict__ W0t, const u16* __restrict__ W1t,
    const float* __restrict__ bias, const float* __restrict__ addsrc,
    float* __restrict__ outf, u16* __restrict__ outb, int nrows)
{
    __shared__ u16 ldsW[2][NFEAT * NFEAT];   // 64 KB: both weight matrices

    int tid = threadIdx.x;
    {   // cooperative LDS fill: 2 x 2048 uint4
        const uint4* s0 = (const uint4*)W0t;
        const uint4* s1 = (const uint4*)W1t;
        uint4* d0 = (uint4*)&ldsW[0][0];
        uint4* d1 = (uint4*)&ldsW[1][0];
        for (int i = tid; i < 2048; i += 256) { d0[i] = s0[i]; d1[i] = s1[i]; }
    }
    __syncthreads();

    int wave = tid >> 5;
    int lane = tid & 31;
    int rowBase = blockIdx.x * 128 + wave * 16;     // wave-uniform -> EXEC all-ones
    if (rowBase >= nrows) return;                   // nrows is a multiple of 16

    int nl   = lane & 15;
    int koff = (lane >> 4) * 8;                     // lanes 16-31 hold the +8 K-halves
    int mhi  = (lane >> 4) * 8;                     // C-tile: lanes 16-31 hold rows M+8

    v8f zero = {0.f, 0.f, 0.f, 0.f, 0.f, 0.f, 0.f, 0.f};
    v8f acc[8];
#pragma unroll
    for (int j = 0; j < 8; ++j) acc[j] = zero;

#pragma unroll
    for (int s = 0; s < 2; ++s) {
        const u16* arow  = (s ? A1 : A0) + (size_t)(rowBase + nl) * NFEAT + koff;
        const u16* lbase = &ldsW[s][(size_t)nl * NFEAT + koff];
#pragma unroll
        for (int k0 = 0; k0 < NFEAT; k0 += 32) {
            // A fragment, ISA 16-bit 16x32 layout: K[k0+koff..+7] and K[k0+koff+16..+23]
            FragU af;
            af.q[0] = *(const uint4*)(arow + k0);
            af.q[1] = *(const uint4*)(arow + k0 + 16);
            // Preload ALL 8 B fragments first so the ds_load clause overlaps the WMMA chain
            FragU bfr[8];
#pragma unroll
            for (int j = 0; j < 8; ++j) {
                const u16* bp = lbase + (size_t)j * 16 * NFEAT + k0;
                bfr[j].q[0] = *(const uint4*)(bp);
                bfr[j].q[1] = *(const uint4*)(bp + 16);
            }
#pragma unroll
            for (int j = 0; j < 8; ++j) {
                acc[j] = __builtin_amdgcn_wmma_f32_16x16x32_bf16(
                    false, af.v, false, bfr[j].v, (short)0, acc[j], false, false);
            }
        }
    }

    // Epilogue per ISA C-layout: VGPR v of tile j -> (row = rowBase + mhi + v, col = 16j + nl)
#pragma unroll
    for (int j = 0; j < 8; ++j) {
        int col = 16 * j + nl;
        float bv = bias[col];
        int r0 = rowBase + mhi;
#pragma unroll
        for (int v = 0; v < 8; ++v) {
            float r = acc[j][v] + bv;
            r = r > 0.0f ? r : 0.0f;
            size_t idx = (size_t)(r0 + v) * NFEAT + col;
            if (AVG) {
                outf[idx] = 0.5f * (addsrc[idx] + r);
            } else {
                outf[idx] = r;
                outb[idx] = f2bf(r);
            }
        }
    }
}

// ---------------- launch ----------------
extern "C" void kernel_launch(void* const* d_in, const int* in_sizes, int n_in,
                              void* d_out, int out_size, void* d_ws, size_t ws_size,
                              hipStream_t stream) {
    const float* x    = (const float*)d_in[0];
    const int*   ei   = (const int*)d_in[1];
    const float* W0_1 = (const float*)d_in[2];
    const float* W1_1 = (const float*)d_in[3];
    const float* b1   = (const float*)d_in[4];
    const float* W0_2 = (const float*)d_in[5];
    const float* W1_2 = (const float*)d_in[6];
    const float* b2   = (const float*)d_in[7];

    const int N  = in_sizes[0] / NFEAT;     // 100000
    const int E  = in_sizes[1] / 2;         // 1600000
    const long NF = (long)N * NFEAT;
    const int* rowp = ei;
    const int* colp = ei + E;

    // workspace carve (256B aligned)
    char* p = (char*)d_ws;
    auto alloc = [&](size_t bytes) { void* r = (void*)p; p += (bytes + 255) & ~(size_t)255; return r; };
    float* deg  = (float*)alloc((size_t)N * 4);
    float* dinv = (float*)alloc((size_t)N * 4);
    u16*   xb   = (u16*)  alloc((size_t)NF * 2);
    float* txf  = (float*)alloc((size_t)NF * 4);
    u16*   txb  = (u16*)  alloc((size_t)NF * 2);
    u16*   h1b  = (u16*)  alloc((size_t)NF * 2);
    u16*   Wt0  = (u16*)  alloc(NFEAT * NFEAT * 2);
    u16*   Wt1  = (u16*)  alloc(NFEAT * NFEAT * 2);
    u16*   Wt2  = (u16*)  alloc(NFEAT * NFEAT * 2);
    u16*   Wt3  = (u16*)  alloc(NFEAT * NFEAT * 2);

    float* out = (float*)d_out;

    const int T = 256;
    int gN   = (N + T - 1) / T;
    int gNF  = (int)((NF + T - 1) / T);
    int gE   = (E + T - 1) / T;
    int gEw  = (E * 32 + T - 1) / T;        // one wave per edge
    int gW   = (NFEAT * NFEAT) / T;
    int gG   = (N + 127) / 128;

    // --- normalization ---
    zero_f32_kernel<<<gN, T, 0, stream>>>(deg, N);
    zero_f32_kernel<<<gNF, T, 0, stream>>>(txf, NF);
    degree_kernel<<<gE, T, 0, stream>>>(rowp, deg, E);
    dinv_kernel<<<gN, T, 0, stream>>>(deg, dinv, N);

    // --- staging converts ---
    cvt_bf16_kernel<<<gNF, T, 0, stream>>>(x, xb, NF);
    wtrans_kernel<<<gW, T, 0, stream>>>(W0_1, Wt0);
    wtrans_kernel<<<gW, T, 0, stream>>>(W1_1, Wt1);
    wtrans_kernel<<<gW, T, 0, stream>>>(W0_2, Wt2);
    wtrans_kernel<<<gW, T, 0, stream>>>(W1_2, Wt3);

    // --- layer 1 ---
    spmm_kernel<<<gEw, T, 0, stream>>>(rowp, colp, dinv, x, txf, E);
    cvt_bf16_kernel<<<gNF, T, 0, stream>>>(txf, txb, NF);
    gemm_dual_kernel<false><<<gG, T, 0, stream>>>(xb, txb, Wt0, Wt1, b1,
                                                  /*addsrc=*/nullptr, out, h1b, N);

    // --- layer 2 (gathers x1 f32 from d_out; fused (x1 + relu)/2 epilogue) ---
    zero_f32_kernel<<<gNF, T, 0, stream>>>(txf, NF);
    spmm_kernel<<<gEw, T, 0, stream>>>(rowp, colp, dinv, out, txf, E);
    cvt_bf16_kernel<<<gNF, T, 0, stream>>>(txf, txb, NF);
    gemm_dual_kernel<true><<<gG, T, 0, stream>>>(h1b, txb, Wt2, Wt3, b2,
                                                 /*addsrc=*/out, out, /*outb=*/nullptr, N);
}
